// AudioGRU_63058709840424
// MI455X (gfx1250) — matrix operand order repaired and checked by reading, not compiled, not run on hardware
//
#include <hip/hip_runtime.h>
#include <hip/hip_bf16.h>

typedef __attribute__((ext_vector_type(16))) __bf16 v16bf;
typedef __attribute__((ext_vector_type(8)))  __bf16 v8bf;
typedef __attribute__((ext_vector_type(8)))  float  v8f;

#define GB 256     // batch
#define GT 2000    // time steps
#define GI 23      // input features
#define GH 128     // hidden
#define TB 16      // batch tile (WMMA M)
#define HS_STRIDE 136   // bf16 elems; 272B = 17*16B -> aligned, bank-skewed
#define XS_STRIDE 40    // bf16 elems; 80B  = 5*16B  -> aligned

__device__ __forceinline__ float sigmoid_f(float x) {
  x = fminf(fmaxf(x, -30.f), 30.f);
  return __fdividef(1.f, 1.f + __expf(-x));
}
__device__ __forceinline__ float tanh_f(float x) {
  x = fminf(fmaxf(x, -15.f), 15.f);
  float e = __expf(-2.f * x);
  return __fdividef(1.f - e, 1.f + e);
}

// A-matrix fragment (16x32 bf16, wave32): lane holds row M=lane&15.
// Elements e=0..15 <-> K = half*16 + e       (two contiguous 16B chunks)
__device__ __forceinline__ v16bf load_a_frag(const __bf16* rowbase, int half) {
  v8bf lo = *(const v8bf*)(rowbase + half * 16);
  v8bf hi = *(const v8bf*)(rowbase + half * 16 + 8);
  v16bf f;
#pragma unroll
  for (int i = 0; i < 8; ++i) { f[i] = lo[i]; f[i + 8] = hi[i]; }
  return f;
}

__launch_bounds__(256, 1)
__global__ void gru_fused_kernel(const float* __restrict__ x,
                                 const float* __restrict__ W_ih,
                                 const float* __restrict__ W_hh,
                                 const float* __restrict__ b_ih,
                                 const float* __restrict__ b_hh,
                                 float* __restrict__ out) {
  // ping-pong buffers -> single barrier per timestep
  __shared__ __align__(16) __bf16 h_s[2][TB * HS_STRIDE];
  __shared__ __align__(16) __bf16 x_s[2][TB * XS_STRIDE];

  const int tid  = threadIdx.x;
  const int lane = tid & 31;
  const int wave = tid >> 5;      // 0..7 -> which 16 H-columns this wave owns
  const int half = lane >> 4;
  const int nloc = lane & 15;     // N (column) within the 16-wide tile
  const int n0   = wave * 16;
  const int b0   = blockIdx.x * TB;

  // ---- one-time: pin weight B-fragments in VGPRs (bf16) ----
  // B-matrix fragment (32x16): lane holds column N=lane&15; element e <-> K = half*16 + e.
  v16bf Bhh[3][4];    // [gate r,z,n][k-chunk of 32]
#pragma unroll
  for (int g = 0; g < 3; ++g) {
    const float* wrow = W_hh + (size_t)(g * GH + n0 + nloc) * GH;
#pragma unroll
    for (int kc = 0; kc < 4; ++kc) {
      const float* p = wrow + kc * 32 + half * 16;
#pragma unroll
      for (int e = 0; e < 16; ++e) Bhh[g][kc][e] = (__bf16)p[e];
    }
  }
  v16bf Bih[3];       // input projection, K padded 23 -> 32
#pragma unroll
  for (int g = 0; g < 3; ++g) {
    const float* wrow = W_ih + (size_t)(g * GH + n0 + nloc) * GI;
#pragma unroll
    for (int e = 0; e < 16; ++e) {
      int k = half * 16 + e;
      Bih[g][e] = (k < GI) ? (__bf16)wrow[k] : (__bf16)0.f;
    }
  }
  // biases: r/z can use pre-summed (b_ih+b_hh); n needs them split
  const int bcol = n0 + nloc;
  const float brz0 = b_ih[0 * GH + bcol] + b_hh[0 * GH + bcol];
  const float brz1 = b_ih[1 * GH + bcol] + b_hh[1 * GH + bcol];
  const float bin2 = b_ih[2 * GH + bcol];
  const float bhn2 = b_hh[2 * GH + bcol];

  // zero both hidden buffers (h0 = 0; buffer 1 is written before first read)
  for (int i = tid; i < TB * HS_STRIDE; i += 256) {
    h_s[0][i] = (__bf16)0.f;
    h_s[1][i] = (__bf16)0.f;
  }

  float hold[8], hsum[8];
#pragma unroll
  for (int v = 0; v < 8; ++v) { hold[v] = 0.f; hsum[v] = 0.f; }

  // cooperative x-tile loading: 16x32 bf16 tile = 512 elems, 2 per thread,
  // register double-buffered (prefetch t+1 while computing t), NT loads.
  const int xr_row0 = tid >> 5, xr_col = tid & 31;
  const int xr_row1 = xr_row0 + 8;
  const size_t xbase0 = (size_t)(b0 + xr_row0) * GT * GI + xr_col;
  const size_t xbase1 = (size_t)(b0 + xr_row1) * GT * GI + xr_col;
  float xr0 = (xr_col < GI) ? __builtin_nontemporal_load(x + xbase0) : 0.f;
  float xr1 = (xr_col < GI) ? __builtin_nontemporal_load(x + xbase1) : 0.f;

  __syncthreads();

  for (int t = 0; t < GT; ++t) {
    const int cur = t & 1, nxt = cur ^ 1;

    // publish this step's x tile into the current buffer
    x_s[cur][xr_row0 * XS_STRIDE + xr_col] = (__bf16)xr0;
    x_s[cur][xr_row1 * XS_STRIDE + xr_col] = (__bf16)xr1;
    // prefetch next step's x (independent of h -> overlaps the WMMA chain)
    if (t + 1 < GT) {
      size_t o = (size_t)(t + 1) * GI;
      xr0 = (xr_col < GI) ? __builtin_nontemporal_load(x + xbase0 + o) : 0.f;
      xr1 = (xr_col < GI) ? __builtin_nontemporal_load(x + xbase1 + o) : 0.f;
    }

    __syncthreads();   // single barrier: x_s[cur] ready, h_s[cur] ready

    // r/z: fused input+recurrent accumulators; n: split (needs r * (gh_n + b))
    v8f acc_r = {}, acc_z = {}, axn = {}, ahn = {};

    {
      v16bf a = load_a_frag(&x_s[cur][nloc * XS_STRIDE], half);
      acc_r = __builtin_amdgcn_wmma_f32_16x16x32_bf16(false, a, false, Bih[0],
                                                      (short)0, acc_r, false, false);
      acc_z = __builtin_amdgcn_wmma_f32_16x16x32_bf16(false, a, false, Bih[1],
                                                      (short)0, acc_z, false, false);
      axn   = __builtin_amdgcn_wmma_f32_16x16x32_bf16(false, a, false, Bih[2],
                                                      (short)0, axn, false, false);
    }
#pragma unroll
    for (int kc = 0; kc < 4; ++kc) {
      v16bf a = load_a_frag(&h_s[cur][nloc * HS_STRIDE + kc * 32], half);
      acc_r = __builtin_amdgcn_wmma_f32_16x16x32_bf16(false, a, false, Bhh[0][kc],
                                                      (short)0, acc_r, false, false);
      acc_z = __builtin_amdgcn_wmma_f32_16x16x32_bf16(false, a, false, Bhh[1][kc],
                                                      (short)0, acc_z, false, false);
      ahn   = __builtin_amdgcn_wmma_f32_16x16x32_bf16(false, a, false, Bhh[2][kc],
                                                      (short)0, ahn, false, false);
    }

    // gates (C/D fragment: element v <-> row M = v + half*8, col = n0+nloc)
#pragma unroll
    for (int v = 0; v < 8; ++v) {
      float r = sigmoid_f(acc_r[v] + brz0);
      float z = sigmoid_f(acc_z[v] + brz1);
      float n = tanh_f(axn[v] + bin2 + r * (ahn[v] + bhn2));
      float h = (1.f - z) * n + z * hold[v];
      hold[v] = h;
      hsum[v] += h;
      int m = v + half * 8;
      h_s[nxt][m * HS_STRIDE + bcol] = (__bf16)h;   // next step's A-matrix
    }
  }

  // time-mean write-out: out[B, H] fp32
#pragma unroll
  for (int v = 0; v < 8; ++v) {
    int m = v + half * 8;
    out[(size_t)(b0 + m) * GH + bcol] = hsum[v] * (1.f / (float)GT);
  }
}

extern "C" void kernel_launch(void* const* d_in, const int* in_sizes, int n_in,
                              void* d_out, int out_size, void* d_ws, size_t ws_size,
                              hipStream_t stream) {
  (void)in_sizes; (void)n_in; (void)d_ws; (void)ws_size; (void)out_size;
  const float* x    = (const float*)d_in[0];
  const float* W_ih = (const float*)d_in[1];
  const float* W_hh = (const float*)d_in[2];
  const float* b_ih = (const float*)d_in[3];
  const float* b_hh = (const float*)d_in[4];
  float* out = (float*)d_out;

  gru_fused_kernel<<<GB / TB, 256, 0, stream>>>(x, W_ih, W_hh, b_ih, b_hh, out);
}
